// EfficientNodeLabelling_652835029803
// MI455X (gfx1250) — matrix operand order repaired and checked by reading, not compiled, not run on hardware
//
#include <hip/hip_runtime.h>
#include <hip/hip_bf16.h>

typedef _Float16 v4h  __attribute__((ext_vector_type(4)));
typedef _Float16 v8h  __attribute__((ext_vector_type(8)));
typedef _Float16 v16h __attribute__((ext_vector_type(16)));
typedef float    v8f  __attribute__((ext_vector_type(8)));

#define H 128

// ---------------------------------------------------------------------------
// Kernel 1: segment boundaries via binary search on sorted batch ids.
// seg[e] = lower_bound(batch, e); seg[E] = T.
// ---------------------------------------------------------------------------
__global__ void segstart_kernel(const long long* __restrict__ batch,
                                int* __restrict__ seg, int T, int E) {
    int e = blockIdx.x * blockDim.x + threadIdx.x;
    if (e > E) return;
    int lo = 0, hi = T;
    long long key = (long long)e;
    while (lo < hi) {
        int mid = (lo + hi) >> 1;
        if (batch[mid] < key) lo = mid + 1; else hi = mid;
    }
    seg[e] = lo;
}

// ---------------------------------------------------------------------------
// Kernel 2: pre-swizzle w1/w2 (f32 row-major KxN, K=N=128) into f16 B-matrix
// fragments for v_wmma_f32_16x16x32_f16.
// Fragment (L, kc, nt): 32 lanes x 16 halfs. Lane l, half j holds
//   W[kc*32 + (j/8)*16 + (l/16)*8 + (j%8)][nt*16 + (l%16)]
// so the GEMM can load each lane's 16 halfs as two contiguous b128s.
// ---------------------------------------------------------------------------
__global__ void pack_weights_kernel(const float* __restrict__ w1,
                                    const float* __restrict__ w2,
                                    _Float16* __restrict__ wpk) {
    int idx = blockIdx.x * blockDim.x + threadIdx.x;   // 0 .. 32767
    int j    = idx & 15;  int r = idx >> 4;
    int lane = r & 31;    r >>= 5;
    int nt   = r & 7;     r >>= 3;
    int kc   = r & 3;     r >>= 2;
    int L    = r & 1;
    int K = kc * 32 + (j >> 3) * 16 + (lane >> 4) * 8 + (j & 7);
    int N = nt * 16 + (lane & 15);
    const float* w = L ? w2 : w1;
    wpk[(((L * 4 + kc) * 8 + nt) * 32 + lane) * 16 + j] = (_Float16)w[K * H + N];
}

// ---------------------------------------------------------------------------
// Kernel 3: fused gather + segment-mean + 3-layer MLP.
// Block = 256 threads = 8 wave32; each wave owns a 16-edge tile.
// ---------------------------------------------------------------------------
__device__ inline v16h load_a_frag(const _Float16* __restrict__ base,
                                   int nl, int h, int kc) {
    // 16-bit A 16x32 layout: lane<16 -> K = (j%8) + (j/8)*16 ; lane>=16 -> +8
    const v8h* p0 = (const v8h*)(base + nl * H + kc * 32 + h * 8);
    const v8h* p1 = (const v8h*)(base + nl * H + kc * 32 + 16 + h * 8);
    v8h lo = *p0, hi = *p1;
    return __builtin_shufflevector(lo, hi, 0, 1, 2, 3, 4, 5, 6, 7,
                                           8, 9, 10, 11, 12, 13, 14, 15);
}

__device__ inline v16h load_b_frag(const _Float16* __restrict__ wpk,
                                   int L, int kc, int nt, int lane) {
    const v8h* bp = (const v8h*)(wpk + (((L * 4 + kc) * 8 + nt) * 32 + lane) * 16);
    v8h lo = bp[0], hi = bp[1];
    return __builtin_shufflevector(lo, hi, 0, 1, 2, 3, 4, 5, 6, 7,
                                           8, 9, 10, 11, 12, 13, 14, 15);
}

__global__ __launch_bounds__(256) void fused_mlp_kernel(
    const long long* __restrict__ z,        // [T,2] int64
    const float* __restrict__ tab_g,        // [4,128]
    const float* __restrict__ b1g,          // [128]
    const float* __restrict__ b2g,          // [128]
    const float* __restrict__ w3g,          // [128]
    const float* __restrict__ b3g,          // [1]
    const int* __restrict__ seg,            // [E+1]
    const _Float16* __restrict__ wpk,       // packed B frags, 2 layers
    float* __restrict__ out)                // [E]
{
    __shared__ float    tab[4 * H];          // embedding table, 2 KB
    __shared__ _Float16 xstage[8 * 16 * H];  // per-wave 16x128 f16 tile, 32 KB

    const int tid  = threadIdx.x;
    const int lane = tid & 31;
    const int wave = tid >> 5;
    const int nl   = lane & 15;   // N within tile / row within tile
    const int h    = lane >> 4;   // half-wave select

    for (int i = tid; i < 4 * H; i += 256) tab[i] = tab_g[i];
    __syncthreads();

    const int tile = blockIdx.x * 8 + wave;
    const int row0 = tile * 16;                       // first edge of tile
    _Float16* xw = xstage + wave * (16 * H);          // per-wave LDS region

    // ---- Phase 1: gather + segment mean -> 16x128 f16 tile in LDS --------
    for (int e16 = 0; e16 < 16; ++e16) {
        const int e  = row0 + e16;
        const int s  = seg[e];
        const int t1 = seg[e + 1];
        __builtin_prefetch(&z[2 * (long long)s], 0, 0);
        float a0 = 0.f, a1 = 0.f, a2 = 0.f, a3 = 0.f;
        for (int t = s; t < t1; ++t) {
            const int z0 = (int)z[2 * t];
            const int z1 = (int)z[2 * t + 1];
            const float* r0 = tab + z0 * H + lane * 4;
            const float* r1 = tab + z1 * H + lane * 4;
            a0 += r0[0] + r1[0];
            a1 += r0[1] + r1[1];
            a2 += r0[2] + r1[2];
            a3 += r0[3] + r1[3];
        }
        const int cnt = t1 - s;
        const float inv = 1.0f / (float)(cnt > 0 ? cnt : 1);
        v4h hv;
        hv[0] = (_Float16)(a0 * inv);
        hv[1] = (_Float16)(a1 * inv);
        hv[2] = (_Float16)(a2 * inv);
        hv[3] = (_Float16)(a3 * inv);
        *(v4h*)(xw + e16 * H + lane * 4) = hv;
    }
    __syncthreads();

    // ---- Phase 2: layer 1  (16x128) @ (128x128) + b1, ReLU ---------------
    v16h A[4];
#pragma unroll
    for (int kc = 0; kc < 4; ++kc) A[kc] = load_a_frag(xw, nl, h, kc);

#pragma unroll
    for (int nt = 0; nt < 8; ++nt) {
        const float bias = b1g[nt * 16 + nl];
        v8f c = {bias, bias, bias, bias, bias, bias, bias, bias};
#pragma unroll
        for (int kc = 0; kc < 4; ++kc) {
            v16h b = load_b_frag(wpk, 0, kc, nt, lane);
            c = __builtin_amdgcn_wmma_f32_16x16x32_f16(
                    false, A[kc], false, b, (short)0, c, false, false);
        }
        // ReLU + restage D (N along lanes) back to row-major LDS for next A
#pragma unroll
        for (int r = 0; r < 8; ++r) {
            float v = c[r] > 0.f ? c[r] : 0.f;
            xw[(r + h * 8) * H + nt * 16 + nl] = (_Float16)v;
        }
    }
    __syncthreads();

    // ---- Phase 3: layer 2  (16x128) @ (128x128) + b2, ReLU ---------------
#pragma unroll
    for (int kc = 0; kc < 4; ++kc) A[kc] = load_a_frag(xw, nl, h, kc);

    v8f acc2[8];
#pragma unroll
    for (int nt = 0; nt < 8; ++nt) {
        const float bias = b2g[nt * 16 + nl];
        v8f c = {bias, bias, bias, bias, bias, bias, bias, bias};
#pragma unroll
        for (int kc = 0; kc < 4; ++kc) {
            v16h b = load_b_frag(wpk, 1, kc, nt, lane);
            c = __builtin_amdgcn_wmma_f32_16x16x32_f16(
                    false, A[kc], false, b, (short)0, c, false, false);
        }
#pragma unroll
        for (int r = 0; r < 8; ++r) c[r] = c[r] > 0.f ? c[r] : 0.f;
        acc2[nt] = c;
    }

    // ---- Phase 4: layer 3  dot with w3 + cross-lane reduction -------------
    float p[8];
#pragma unroll
    for (int r = 0; r < 8; ++r) p[r] = 0.f;
#pragma unroll
    for (int nt = 0; nt < 8; ++nt) {
        const float wv = w3g[nt * 16 + nl];
#pragma unroll
        for (int r = 0; r < 8; ++r) p[r] += acc2[nt][r] * wv;
    }
    const float bias3 = b3g[0];
#pragma unroll
    for (int r = 0; r < 8; ++r) {
        float s = p[r];
        // reduce across the 16 lanes sharing one matrix row (halves disjoint)
        s += __shfl_xor(s, 1, 32);
        s += __shfl_xor(s, 2, 32);
        s += __shfl_xor(s, 4, 32);
        s += __shfl_xor(s, 8, 32);
        if (nl == 0) {
            out[row0 + h * 8 + r] = s + bias3;   // lane 0 -> M=r, lane 16 -> M=r+8
        }
    }
}

// ---------------------------------------------------------------------------
extern "C" void kernel_launch(void* const* d_in, const int* in_sizes, int n_in,
                              void* d_out, int out_size, void* d_ws, size_t ws_size,
                              hipStream_t stream) {
    const long long* z     = (const long long*)d_in[0];
    const long long* batch = (const long long*)d_in[1];
    const float*     tab   = (const float*)d_in[2];
    const float*     w1    = (const float*)d_in[3];
    const float*     b1    = (const float*)d_in[4];
    const float*     w2    = (const float*)d_in[5];
    const float*     b2    = (const float*)d_in[6];
    const float*     w3    = (const float*)d_in[7];
    const float*     b3    = (const float*)d_in[8];
    float* out = (float*)d_out;

    const int T = in_sizes[1];   // batch has T elements
    const int E = out_size;      // one logit per edge

    // workspace layout: [ seg : (E+1) ints ][ align 256 ][ wpk : 2*16384 f16 ]
    int* seg = (int*)d_ws;
    size_t wpk_off = (((size_t)(E + 1) * 4) + 255) & ~(size_t)255;
    _Float16* wpk = (_Float16*)((char*)d_ws + wpk_off);

    pack_weights_kernel<<<128, 256, 0, stream>>>(w1, w2, wpk);
    segstart_kernel<<<(E + 1 + 255) / 256, 256, 0, stream>>>(batch, seg, T, E);
    fused_mlp_kernel<<<E / 128, 256, 0, stream>>>(z, tab, b1, b2, w3, b3,
                                                  seg, wpk, out);
}
